// BasicRGCN_24283745092172
// MI455X (gfx1250) — matrix operand order repaired and checked by reading, not compiled, not run on hardware
//
#include <hip/hip_runtime.h>

#define RNUM 4
#define NN   4096
#define FF   64

typedef __attribute__((ext_vector_type(16))) __bf16 v16bf;
typedef __attribute__((ext_vector_type(8)))  __bf16 v8bf;
typedef __attribute__((ext_vector_type(8)))  float  v8f;

__global__ void zero_f32(float* __restrict__ p, int n) {
    int i = blockIdx.x * blockDim.x + threadIdx.x;
    if (i < n) p[i] = 0.0f;
}

// xwT[r][g][m] = sum_f x[m][f] * W[r][g][f]   (W pre-offset to the layer), bf16 out
__global__ void xwT_kernel(const float* __restrict__ x,
                           const float* __restrict__ W,
                           __bf16* __restrict__ xwT) {
    int idx = blockIdx.x * blockDim.x + threadIdx.x;   // R*F*N = 1M threads
    int m = idx & (NN - 1);
    int g = (idx >> 12) & (FF - 1);
    int r = idx >> 18;
    const float* xr = x + (size_t)m * FF;
    const float* wr = W + ((size_t)(r * FF + g)) * FF;
    float acc = 0.0f;
    #pragma unroll
    for (int f = 0; f < FF; ++f) acc += xr[f] * wr[f];
    xwT[idx] = (__bf16)acc;     // idx == (r*FF+g)*NN + m
}

// xm[r][n][g] = sum_f x[n][f] * M[r][f][g], bf16 out
__global__ void xm_kernel(const float* __restrict__ x,
                          const float* __restrict__ Mrel,
                          __bf16* __restrict__ xm) {
    int idx = blockIdx.x * blockDim.x + threadIdx.x;   // R*N*F = 1M threads
    int g = idx & (FF - 1);
    int n = (idx >> 6) & (NN - 1);
    int r = idx >> 18;
    const float* xr = x + (size_t)n * FF;
    const float* mr = Mrel + (size_t)r * FF * FF + g;
    float acc = 0.0f;
    #pragma unroll
    for (int f = 0; f < FF; ++f) acc += xr[f] * mr[(size_t)f * FF];
    xm[idx] = (__bf16)acc;      // idx == (r*NN+n)*FF + g
}

__global__ void sigmoid_k(const float* __restrict__ y,
                          float* __restrict__ xA,
                          __bf16* __restrict__ xb, int n) {
    int i = blockIdx.x * blockDim.x + threadIdx.x;
    if (i < n) {
        float s = 1.0f / (1.0f + __expf(-y[i]));
        xA[i] = s;
        xb[i] = (__bf16)s;
    }
}

// y[n][0..63] += sum over one (r, m-chunk) of A[r,n,m] * xwT[r,g,m]
// Wave per (n_tile 256) x (r 4) x (k_chunk 8) = 8192 waves; each wave computes a
// full 16x64 strip so every A-fragment (adjacency bytes + f32->bf16 cvt) feeds
// 4 WMMAs. Adjacency is streamed exactly once per layer.
__global__ void conv_bf16(const float* __restrict__ A,    // [R,N,N] f32
                          const __bf16* __restrict__ xwT, // [R,F,N] bf16
                          float* __restrict__ y) {        // [N,F] f32 (atomic acc)
    int wid  = (blockIdx.x << 3) + (threadIdx.x >> 5);
    int lane = threadIdx.x & 31;
    int nt = wid & 255;
    int r  = (wid >> 8) & 3;
    int kc =  wid >> 10;                       // 0..7, chunks of 512 over m
    int n0 = nt << 4;
    int lrow = lane & 15, hi = lane >> 4, roff = hi << 3;

    const float*  Arow = A + ((size_t)r * NN + (n0 + lrow)) * NN + (kc << 9);
    // B base for g-tile 0; g-tiles are 16*NN bf16 elements apart
    const __bf16* Bcol = xwT + ((size_t)(r * FF + lrow) * NN) + (kc << 9) + (hi << 4);

    v8f c0 = {}, c1 = {}, c2 = {}, c3 = {};
    for (int t = 0; t < 16; ++t) {
        int k0 = t << 5;
        // A-fragment: bf16 16x32 layout — lane holds K = hi*16..+15, contiguous
        const float* p1 = Arow + k0 + (hi << 4);
        float4 f0 = *(const float4*)(p1);
        float4 f1 = *(const float4*)(p1 + 4);
        float4 f2 = *(const float4*)(p1 + 8);
        float4 f3 = *(const float4*)(p1 + 12);
        v16bf a;
        a[0]=(__bf16)f0.x; a[1]=(__bf16)f0.y; a[2]=(__bf16)f0.z; a[3]=(__bf16)f0.w;
        a[4]=(__bf16)f1.x; a[5]=(__bf16)f1.y; a[6]=(__bf16)f1.z; a[7]=(__bf16)f1.w;
        a[8]=(__bf16)f2.x; a[9]=(__bf16)f2.y; a[10]=(__bf16)f2.z; a[11]=(__bf16)f2.w;
        a[12]=(__bf16)f3.x; a[13]=(__bf16)f3.y; a[14]=(__bf16)f3.z; a[15]=(__bf16)f3.w;
        // B-fragments: 32x16, lane = column (g), elements = K hi*16..+15, contiguous
        v16bf b0 = *(const v16bf*)(Bcol + k0);
        v16bf b1 = *(const v16bf*)(Bcol + (size_t)16 * NN + k0);
        v16bf b2 = *(const v16bf*)(Bcol + (size_t)32 * NN + k0);
        v16bf b3 = *(const v16bf*)(Bcol + (size_t)48 * NN + k0);
        c0 = __builtin_amdgcn_wmma_f32_16x16x32_bf16(false, a, false, b0, (short)0, c0, false, false);
        c1 = __builtin_amdgcn_wmma_f32_16x16x32_bf16(false, a, false, b1, (short)0, c1, false, false);
        c2 = __builtin_amdgcn_wmma_f32_16x16x32_bf16(false, a, false, b2, (short)0, c2, false, false);
        c3 = __builtin_amdgcn_wmma_f32_16x16x32_bf16(false, a, false, b3, (short)0, c3, false, false);
    }
    float* yb = y + (size_t)(n0 + roff) * FF + lrow;
    #pragma unroll
    for (int v = 0; v < 8; ++v) {
        (void)__hip_atomic_fetch_add(yb + (size_t)v * FF,      c0[v], __ATOMIC_RELAXED, __HIP_MEMORY_SCOPE_AGENT);
        (void)__hip_atomic_fetch_add(yb + (size_t)v * FF + 16, c1[v], __ATOMIC_RELAXED, __HIP_MEMORY_SCOPE_AGENT);
        (void)__hip_atomic_fetch_add(yb + (size_t)v * FF + 32, c2[v], __ATOMIC_RELAXED, __HIP_MEMORY_SCOPE_AGENT);
        (void)__hip_atomic_fetch_add(yb + (size_t)v * FF + 48, c3[v], __ATOMIC_RELAXED, __HIP_MEMORY_SCOPE_AGENT);
    }
}

// out[r][n][m] = sum_g xm[r][n][g] * x[m][g]; wave per 16x16 tile (262144 waves).
// Operands are small (L2-resident); traffic is dominated by the streamed output.
__global__ void distmult_bf16(const __bf16* __restrict__ xm, // [R,N,F] bf16
                              const __bf16* __restrict__ xb, // [N,F]  bf16
                              float* __restrict__ out) {     // [R,N,N] f32
    int wid  = (blockIdx.x << 3) + (threadIdx.x >> 5);
    int lane = threadIdx.x & 31;
    int mt = wid & 255;
    int nt = (wid >> 8) & 255;
    int r  =  wid >> 16;
    int n0 = nt << 4, m0 = mt << 4;
    int lrow = lane & 15, hi = lane >> 4, roff = hi << 3;

    const __bf16* Ap = xm + (((size_t)r * NN + n0 + lrow) << 6);
    const __bf16* Bp = xb + (((size_t)(m0 + lrow)) << 6) + (hi << 4);

    v8f c = {};
    #pragma unroll
    for (int t = 0; t < 2; ++t) {
        int k0 = t << 5;
        v8bf alo = *(const v8bf*)(Ap + k0 + (hi << 4));
        v8bf ahi = *(const v8bf*)(Ap + k0 + 8 + (hi << 4));
        v16bf a;
        #pragma unroll
        for (int i = 0; i < 8; ++i) { a[i] = alo[i]; a[8 + i] = ahi[i]; }
        v16bf b = *(const v16bf*)(Bp + k0);
        c = __builtin_amdgcn_wmma_f32_16x16x32_bf16(false, a, false, b,
                                                    (short)0, c, false, false);
    }
    float* ob = out + ((size_t)r * NN + n0 + roff) * NN + m0 + lrow;
    #pragma unroll
    for (int v = 0; v < 8; ++v) ob[(size_t)v * NN] = c[v];
}

extern "C" void kernel_launch(void* const* d_in, const int* in_sizes, int n_in,
                              void* d_out, int out_size, void* d_ws, size_t ws_size,
                              hipStream_t stream) {
    const float* adj  = (const float*)d_in[0];   // [R,N,N]
    const float* feat = (const float*)d_in[1];   // [N,F]
    const float* cw   = (const float*)d_in[2];   // [L,R,F,F]
    const float* rel  = (const float*)d_in[3];   // [R,F,F]
    float* out = (float*)d_out;

    char* ws = (char*)d_ws;
    float*  y   = (float*)(ws);                         // 1.0 MB  [N,F] f32
    float*  xA  = (float*)(ws + (1u << 20));            // 1.0 MB  [N,F] f32
    __bf16* xb  = (__bf16*)(ws + (2u << 20));           // 0.5 MB  [N,F] bf16
    __bf16* xwT = (__bf16*)(ws + (5u << 19));           // 2.0 MB  [R,F,N] bf16
    __bf16* xm  = (__bf16*)(ws + (9u << 19));           // 2.0 MB  [R,N,F] bf16

    const int NE = NN * FF;                    // 262144
    dim3 b256(256);
    const int smallBlocks = (RNUM * FF * NN) / 256;   // 4096
    const int elemBlocks  = NE / 256;                 // 1024
    const int convBlocks  = 8192 / 8;                 // 1024 (256 nt * 4 r * 8 kc)
    const int dmBlocks    = (RNUM * 256 * 256) / 8;   // 32768

    // ----- layer 0 -----
    xwT_kernel<<<smallBlocks, b256, 0, stream>>>(feat, cw, xwT);
    zero_f32  <<<elemBlocks, b256, 0, stream>>>(y, NE);
    conv_bf16 <<<convBlocks, b256, 0, stream>>>(adj, xwT, y);
    sigmoid_k <<<elemBlocks, b256, 0, stream>>>(y, xA, xb, NE);
    // ----- layer 1 -----
    xwT_kernel<<<smallBlocks, b256, 0, stream>>>(xA, cw + (size_t)RNUM * FF * FF, xwT);
    zero_f32  <<<elemBlocks, b256, 0, stream>>>(y, NE);
    conv_bf16 <<<convBlocks, b256, 0, stream>>>(adj, xwT, y);
    sigmoid_k <<<elemBlocks, b256, 0, stream>>>(y, xA, xb, NE);
    // ----- DistMult -----
    xm_kernel <<<smallBlocks, b256, 0, stream>>>(xA, rel, xm);
    distmult_bf16<<<dmBlocks, b256, 0, stream>>>(xm, xb, out);
}
